// CNV_48498770706530
// MI455X (gfx1250) — 8-bit matrix operand order repaired and checked by reading, compiled, not run on hardware
//
#include <hip/hip_runtime.h>

typedef signed char s8;
typedef __attribute__((ext_vector_type(8))) int v8i;
typedef __attribute__((ext_vector_type(4))) int v4i;

#define EPSV 1e-4f

#if defined(__has_builtin)
#if __has_builtin(__builtin_amdgcn_global_load_async_to_lds_b128)
#define ASYNC_LDS 1
#endif
#endif
#ifndef ASYNC_LDS
#define ASYNC_LDS 0
#endif

typedef __attribute__((address_space(1))) v4i g_v4i;
typedef __attribute__((address_space(3))) v4i l_v4i;

__device__ __forceinline__ void copy16_to_lds(const s8* __restrict__ g,
                                              s8* __restrict__ l) {
#if ASYNC_LDS
  __builtin_amdgcn_global_load_async_to_lds_b128((g_v4i*)g, (l_v4i*)l, 0, 0);
#else
  *(int4*)l = *(const int4*)g;
#endif
}

__device__ __forceinline__ void wait_async_copies() {
#if ASYNC_LDS
#if __has_builtin(__builtin_amdgcn_s_wait_asynccnt)
  __builtin_amdgcn_s_wait_asynccnt(0);
#else
  asm volatile("s_wait_asynccnt 0" ::: "memory");
#endif
#endif
}


__device__ __forceinline__ v8i frag_a_lds(const s8* lds, int mOff, int lane) {
  v8i f;
  const s8* row = lds + (mOff + (lane & 15)) * 80;
  const int hk = (lane >> 4) << 3;
#pragma unroll
  for (int v = 0; v < 8; ++v) {
    int kb = ((v >> 1) << 4) + ((v & 1) << 2) + hk;
    f[v] = *(const int*)(row + kb);
  }
  return f;
}

__device__ __forceinline__ v8i frag_b_lds(const s8* lds, int nOff, int lane) {
  v8i f;
  const s8* row = lds + (nOff + (lane & 15)) * 80;
  const int hk = (lane >> 4) << 3;
#pragma unroll
  for (int v = 0; v < 8; ++v) {
    int kb = ((v >> 1) << 4) + ((v & 1) << 2) + hk;
    f[v] = *(const int*)(row + kb);
  }
  return f;
}

template <bool IS_FC>
__global__ __launch_bounds__(128, 4)
void gemm_tern(const s8* __restrict__ A, const s8* __restrict__ Bt,
               const float* __restrict__ sA, const float* __restrict__ bA,
               s8* __restrict__ Out, int M, int K, int N,
               int Hin, int Win, int Hout, int Wout, int Cin) {
  __shared__ __align__(16) s8 ldsA[2][64 * 80];
  __shared__ __align__(16) s8 ldsB[2][64 * 80];

  const int tid  = threadIdx.x;
  const int lane = tid & 31;
  const int wv   = tid >> 5;
  const int m0   = blockIdx.x * 64;
  const int n0   = blockIdx.y * 64;

  const int t = tid >> 1;
  const int h = tid & 1;

  const s8* rowBaseA;
  if (IS_FC) {
    rowBaseA = A + (size_t)(m0 + t) * (size_t)K + (h << 5);
  } else {
    const int HW  = Hout * Wout;
    const int R   = m0 + t;
    const int n_  = R / HW;
    const int rem = R - n_ * HW;
    const int oy  = rem / Wout;
    const int ox  = rem - oy * Wout;
    rowBaseA = A + (((size_t)n_ * Hin + oy) * Win + ox) * (size_t)Cin + (h << 5);
  }
  const s8* rowBaseB = Bt + (size_t)(n0 + t) * (size_t)K + (h << 5);

  s8* const stA0 = &ldsA[0][t * 80 + (h << 5)];
  s8* const stA1 = &ldsA[1][t * 80 + (h << 5)];
  s8* const stB0 = &ldsB[0][t * 80 + (h << 5)];
  s8* const stB1 = &ldsB[1][t * 80 + (h << 5)];

  const v8i vzero = {0, 0, 0, 0, 0, 0, 0, 0};
  v8i acc[4] = {vzero, vzero, vzero, vzero};

  const int Ks = K >> 6;

  int aOffN = 0;
  int c0n = 0, dxn = 0;
  int kkN = 0;

  copy16_to_lds(rowBaseA + aOffN,      stA0);
  copy16_to_lds(rowBaseA + aOffN + 16, stA0 + 16);
  copy16_to_lds(rowBaseB + kkN,        stB0);
  copy16_to_lds(rowBaseB + kkN + 16,   stB0 + 16);
  if (IS_FC) {
    aOffN += 64;
  } else {
    aOffN += 64; c0n += 64;
    if (c0n == Cin) {
      c0n = 0;
      if (++dxn == 3) { dxn = 0; aOffN += (Win - 3) * Cin; }
    }
  }
  kkN += 64;
  wait_async_copies();
  __syncthreads();

  for (int ks = 0; ks < Ks; ++ks) {
    const int cur = ks & 1;

    if (ks + 1 < Ks) {
      s8* dA = cur ? stA0 : stA1;
      s8* dB = cur ? stB0 : stB1;
      copy16_to_lds(rowBaseA + aOffN,      dA);
      copy16_to_lds(rowBaseA + aOffN + 16, dA + 16);
      copy16_to_lds(rowBaseB + kkN,        dB);
      copy16_to_lds(rowBaseB + kkN + 16,   dB + 16);
      if (IS_FC) {
        aOffN += 64;
      } else {
        aOffN += 64; c0n += 64;
        if (c0n == Cin) {
          c0n = 0;
          if (++dxn == 3) { dxn = 0; aOffN += (Win - 3) * Cin; }
        }
      }
      kkN += 64;
    }

    const v8i af = frag_a_lds(ldsA[cur], wv << 4, lane);
#pragma unroll
    for (int nb = 0; nb < 4; ++nb) {
      const v8i bf = frag_b_lds(ldsB[cur], nb << 4, lane);
      acc[nb] = __builtin_amdgcn_wmma_i32_16x16x64_iu8(
           true, af,  true, bf, acc[nb],
           false,  false);
    }

    wait_async_copies();
    __syncthreads();
  }

  const int Rbase = m0 + (wv << 4) + ((lane >> 4) << 3);
  size_t obase[8];
  if (IS_FC) {
#pragma unroll
    for (int r = 0; r < 8; ++r) obase[r] = (size_t)(Rbase + r) * (size_t)N;
  } else {
    const int HW = Hout * Wout;
#pragma unroll
    for (int r = 0; r < 8; ++r) {
      const int R   = Rbase + r;
      const int n_  = R / HW;
      const int rem = R - n_ * HW;
      const int oy  = rem / Wout;
      const int ox  = rem - oy * Wout;
      obase[r] = (((size_t)n_ * Hout + oy) * Wout + ox) * (size_t)N;
    }
  }
#pragma unroll
  for (int nb = 0; nb < 4; ++nb) {
    const int ch = n0 + (nb << 4) + (lane & 15);
    const float s = sA[ch];
    const float b = bA[ch];
#pragma unroll
    for (int r = 0; r < 8; ++r) {
      const float z = (float)acc[nb][r] * s + b;
      const float q = fminf(1.f, fmaxf(-1.f, rintf(z)));
      Out[obase[r] + ch] = (s8)q;
    }
  }
}

__global__ __launch_bounds__(128)
void conv0_kernel(const float* __restrict__ x, const s8* __restrict__ Bt0,
                  const float* __restrict__ sA, const float* __restrict__ bA,
                  s8* __restrict__ out) {
  __shared__ s8 wl[64 * 27];
  for (int i = threadIdx.x; i < 64 * 27; i += 128) wl[i] = Bt0[i];
  __syncthreads();

  const int gid = blockIdx.x * 128 + threadIdx.x;
  const int pix = gid >> 2;
  const int cg  = (gid & 3) << 4;
  const int n   = pix / 900;
  const int rem = pix - n * 900;
  const int oy  = rem / 30;
  const int ox  = rem - oy * 30;

  float iq[27];
#pragma unroll
  for (int dy = 0; dy < 3; ++dy)
#pragma unroll
    for (int dx = 0; dx < 3; ++dx)
#pragma unroll
      for (int c = 0; c < 3; ++c) {
        const float v = x[((n * 3 + c) * 32 + (oy + dy)) * 32 + (ox + dx)];
        iq[(dy * 3 + dx) * 3 + c] =
            fminf(127.f, fmaxf(-128.f, rintf(v * 128.f)));
      }

  s8* op = out + (((size_t)n * 30 + oy) * 30 + ox) * 64;
#pragma unroll 4
  for (int co = cg; co < cg + 16; ++co) {
    float acc = 0.f;
    const s8* wr = wl + co * 27;
#pragma unroll
    for (int k = 0; k < 27; ++k) acc += iq[k] * (float)wr[k];
    const float z = acc * sA[co] + bA[co];
    op[co] = (s8)fminf(1.f, fmaxf(-1.f, rintf(z)));
  }
}

__device__ __forceinline__ s8 pool_tern(int s) {
  return (s8)((s >= 3) ? 1 : ((s <= -3) ? -1 : 0));
}

__global__ void pool_kernel(const s8* __restrict__ in, s8* __restrict__ out,
                            int Nb, int H, int W, int C) {
  const int C4 = C >> 2, Ho = H >> 1, Wo = W >> 1;
  const int total = Nb * Ho * Wo * C4;
  const int gid = blockIdx.x * 256 + threadIdx.x;
  if (gid >= total) return;
  int tv = gid;
  const int c4 = tv % C4; tv /= C4;
  const int xo = tv % Wo; tv /= Wo;
  const int yo = tv % Ho;
  const int n  = tv / Ho;

  const char4* in4 = (const char4*)in;
  const size_t rs  = (size_t)W * C4;
  const size_t b0  = (((size_t)n * H + yo * 2) * W + xo * 2) * C4 + c4;
  const char4 p00 = in4[b0], p01 = in4[b0 + C4];
  const char4 p10 = in4[b0 + rs], p11 = in4[b0 + rs + C4];
  char4 o;
  o.x = pool_tern((int)p00.x + p01.x + p10.x + p11.x);
  o.y = pool_tern((int)p00.y + p01.y + p10.y + p11.y);
  o.z = pool_tern((int)p00.z + p01.z + p10.z + p11.z);
  o.w = pool_tern((int)p00.w + p01.w + p10.w + p11.w);
  ((char4*)out)[(((size_t)n * Ho + yo) * Wo + xo) * C4 + c4] = o;
}

__global__ __launch_bounds__(128)
void fc2_kernel(const s8* __restrict__ A, const s8* __restrict__ Bt2,
                const float* __restrict__ tn, float* __restrict__ out) {
  __shared__ s8 bl[10 * 512];
  for (int i = threadIdx.x; i < 10 * 512; i += 128) bl[i] = Bt2[i];
  __syncthreads();
  const int gid = blockIdx.x * 128 + threadIdx.x;
  if (gid >= 1024 * 10) return;
  const int row = gid / 10, j = gid - row * 10;
  const s8* a = A + (size_t)row * 512;
  const s8* b = bl + j * 512;
  int acc = 0;
#pragma unroll 8
  for (int k = 0; k < 512; ++k) acc += (int)a[k] * (int)b[k];
  const float w_t = tn[0], b_t = tn[1], m_t = tn[2], v_t = tn[3];
  out[gid] = ((float)acc - m_t) * (w_t * rsqrtf(v_t + EPSV)) + b_t;
}

__global__ void pack_convw(const float* __restrict__ w, s8* __restrict__ Bt,
                           int Co, int Ci) {
  const int total = Co * Ci * 9;
  const int idx = blockIdx.x * 256 + threadIdx.x;
  if (idx >= total) return;
  const int co = idx / (Ci * 9);
  const int r  = idx - co * (Ci * 9);
  const int c  = r / 9;
  const int dd = r - c * 9;
  const float v = w[((size_t)(co * Ci + c) * 3 + dd / 3) * 3 + (dd % 3)];
  Bt[(size_t)co * (9 * Ci) + dd * Ci + c] =
      (s8)fminf(1.f, fmaxf(-1.f, rintf(v)));
}

__global__ void pack_fcw(const float* __restrict__ w, s8* __restrict__ Bt,
                         int total) {
  const int idx = blockIdx.x * 256 + threadIdx.x;
  if (idx >= total) return;
  Bt[idx] = (s8)fminf(1.f, fmaxf(-1.f, rintf(w[idx])));
}

__global__ void fold_bn(const float* __restrict__ bn, float* __restrict__ sA,
                        float* __restrict__ bA, int C, float s) {
  const int c = blockIdx.x * 256 + threadIdx.x;
  if (c >= C) return;
  const float g = bn[c], b = bn[C + c], m = bn[2 * C + c], v = bn[3 * C + c];
  const float inv = g * rsqrtf(v + EPSV);
  sA[c] = s * inv;
  bA[c] = b - m * inv;
}

extern "C" void kernel_launch(void* const* d_in, const int* in_sizes, int n_in,
                              void* d_out, int out_size, void* d_ws,
                              size_t ws_size, hipStream_t stream) {
  (void)in_sizes; (void)n_in; (void)out_size; (void)ws_size;
  const float* x = (const float*)d_in[0];
  const float* wc[6];
  const float* bnc[6];
  for (int i = 0; i < 6; ++i) {
    wc[i]  = (const float*)d_in[1 + 2 * i];
    bnc[i] = (const float*)d_in[2 + 2 * i];
  }
  const float* wf0  = (const float*)d_in[13];
  const float* wf1  = (const float*)d_in[14];
  const float* wf2  = (const float*)d_in[15];
  const float* bnf0 = (const float*)d_in[16];
  const float* bnf1 = (const float*)d_in[17];
  const float* tn   = (const float*)d_in[18];

  char* ws = (char*)d_ws;
  size_t off = 0;
  auto carve = [&](size_t n) -> char* {
    char* p = ws + off;
    off += (n + 255) & ~(size_t)255;
    return p;
  };
  s8* bufA = (s8*)carve((size_t)1024 * 30 * 30 * 64);
  s8* bufB = (s8*)carve((size_t)1024 * 28 * 28 * 64);

  const int convCi[6] = {3, 64, 64, 128, 128, 256};
  const int convCo[6] = {64, 64, 128, 128, 256, 256};
  s8* btc[6];
  for (int i = 0; i < 6; ++i)
    btc[i] = (s8*)carve((size_t)convCo[i] * 9 * convCi[i]);
  s8* btf0 = (s8*)carve(512 * 256);
  s8* btf1 = (s8*)carve(512 * 512);
  s8* btf2 = (s8*)carve(10 * 512);
  float* sc[8];
  float* bi[8];
  const int chs[8] = {64, 64, 128, 128, 256, 256, 512, 512};
  for (int i = 0; i < 8; ++i) {
    sc[i] = (float*)carve(chs[i] * 4);
    bi[i] = (float*)carve(chs[i] * 4);
  }

  for (int i = 0; i < 6; ++i) {
    const int tot = convCo[i] * convCi[i] * 9;
    pack_convw<<<(tot + 255) / 256, 256, 0, stream>>>(wc[i], btc[i], convCo[i],
                                                      convCi[i]);
  }
  pack_fcw<<<(512 * 256 + 255) / 256, 256, 0, stream>>>(wf0, btf0, 512 * 256);
  pack_fcw<<<(512 * 512 + 255) / 256, 256, 0, stream>>>(wf1, btf1, 512 * 512);
  pack_fcw<<<(10 * 512 + 255) / 256, 256, 0, stream>>>(wf2, btf2, 10 * 512);
  fold_bn<<<1, 256, 0, stream>>>(bnc[0], sc[0], bi[0], 64, 1.f / 128.f);
  for (int i = 1; i < 6; ++i)
    fold_bn<<<(chs[i] + 255) / 256, 256, 0, stream>>>(bnc[i], sc[i], bi[i],
                                                      chs[i], 1.f);
  fold_bn<<<2, 256, 0, stream>>>(bnf0, sc[6], bi[6], 512, 1.f);
  fold_bn<<<2, 256, 0, stream>>>(bnf1, sc[7], bi[7], 512, 1.f);

  conv0_kernel<<<28800, 128, 0, stream>>>(x, btc[0], sc[0], bi[0], bufA);

  gemm_tern<false><<<dim3(802816 / 64, 1), 128, 0, stream>>>(
      bufA, btc[1], sc[1], bi[1], bufB, 802816, 576, 64, 30, 30, 28, 28, 64);
  {
    const int tot = 1024 * 14 * 14 * 16;
    pool_kernel<<<(tot + 255) / 256, 256, 0, stream>>>(bufB, bufA, 1024, 28, 28,
                                                       64);
  }
  gemm_tern<false><<<dim3(147456 / 64, 2), 128, 0, stream>>>(
      bufA, btc[2], sc[2], bi[2], bufB, 147456, 576, 128, 14, 14, 12, 12, 64);
  gemm_tern<false><<<dim3(102400 / 64, 2), 128, 0, stream>>>(
      bufB, btc[3], sc[3], bi[3], bufA, 102400, 1152, 128, 12, 12, 10, 10, 128);
  {
    const int tot = 1024 * 5 * 5 * 32;
    pool_kernel<<<(tot + 255) / 256, 256, 0, stream>>>(bufA, bufB, 1024, 10, 10,
                                                       128);
  }
  gemm_tern<false><<<dim3(9216 / 64, 4), 128, 0, stream>>>(
      bufB, btc[4], sc[4], bi[4], bufA, 9216, 1152, 256, 5, 5, 3, 3, 128);
  gemm_tern<false><<<dim3(1024 / 64, 4), 128, 0, stream>>>(
      bufA, btc[5], sc[5], bi[5], bufB, 1024, 2304, 256, 3, 3, 1, 1, 256);
  gemm_tern<true><<<dim3(16, 8), 128, 0, stream>>>(
      bufB, btf0, sc[6], bi[6], bufA, 1024, 256, 512, 0, 0, 0, 0, 0);
  gemm_tern<true><<<dim3(16, 8), 128, 0, stream>>>(
      bufA, btf1, sc[7], bi[7], bufB, 1024, 512, 512, 0, 0, 0, 0, 0);
  fc2_kernel<<<80, 128, 0, stream>>>(bufB, btf2, tn, (float*)d_out);
}
